// AttentionLayer_10350871183816
// MI455X (gfx1250) — matrix operand order repaired and checked by reading, not compiled, not run on hardware
//
#include <hip/hip_runtime.h>
#include <hip/hip_bf16.h>

// ---------- types ----------
typedef __attribute__((ext_vector_type(4)))  float  v4f;
typedef __attribute__((ext_vector_type(8)))  float  v8f;
typedef __attribute__((ext_vector_type(4)))  __bf16 v4bf;
typedef __attribute__((ext_vector_type(8)))  __bf16 v8bf;
typedef __attribute__((ext_vector_type(16))) __bf16 v16bf;

#define DMODEL 512
#define NH     8
#define DKH    64
#define LSEQ   1024

// ---------- WMMA fragment helpers (CDNA5 16x16x32 bf16 layout) ----------
// A (16x32 bf16): lane l holds row m=l%16; VGPR0-3 = k in [h2*8, h2*8+8),
// VGPR4-7 = k in [16+h2*8, 16+h2*8+8) where h2 = l/16. Two contiguous 16B runs.
static __device__ __forceinline__ v16bf load_afrag(const __bf16* rowbase, int lane) {
    const int h2 = lane >> 4;
    v16bf a;
    ((v8bf*)&a)[0] = *(const v8bf*)(rowbase + h2 * 8);
    ((v8bf*)&a)[1] = *(const v8bf*)(rowbase + 16 + h2 * 8);
    return a;
}
// B (32x16 bf16): lane l holds column n=l%16; k order same as A (h2*8+s, 16+h2*8+s).
static __device__ __forceinline__ v16bf load_bfrag(const __bf16* base, int ldb, int lane) {
    const int h2 = lane >> 4;
    const __bf16* p = base + (lane & 15);
    v16bf b;
#pragma unroll
    for (int s = 0; s < 8; ++s) {
        b[s]     = p[(size_t)(h2 * 8 + s) * ldb];
        b[8 + s] = p[(size_t)(16 + h2 * 8 + s) * ldb];
    }
    return b;
}
static __device__ __forceinline__ v8f wmma_bf16(v16bf a, v16bf b, v8f c) {
    return __builtin_amdgcn_wmma_f32_16x16x32_bf16(false, a, false, b, (short)0, c,
                                                   false, false);
}

// ---------- fp32 -> bf16 conversion ----------
__global__ void cvt_f32_bf16(const float* __restrict__ in, __bf16* __restrict__ out,
                             long n4) {
    long i = (long)blockIdx.x * blockDim.x + threadIdx.x;
    const long stride = (long)gridDim.x * blockDim.x;
    for (; i < n4; i += stride) {
        v4f f = ((const v4f*)in)[i];
        v4bf o;
        o[0] = (__bf16)f[0]; o[1] = (__bf16)f[1];
        o[2] = (__bf16)f[2]; o[3] = (__bf16)f[3];
        ((v4bf*)out)[i] = o;
    }
}

// ---------- generic bf16 GEMM: C = A[MxK] * B[KxN] + bias ----------
// MODE 0: bf16 out, row-major [M][N]
// MODE 1: bf16 out, K-transposed layout [bt][h][dk][s]  (for QK^T B operand)
// MODE 2: fp32 out, row-major [M][N]                    (final output proj)
template <int MODE>
__global__ void __launch_bounds__(256) gemm_bf16_k(const __bf16* __restrict__ A,
                                                   const __bf16* __restrict__ Bw,
                                                   const float* __restrict__ bias,
                                                   void* __restrict__ OutP,
                                                   int M, int N, int K) {
    const int lane = threadIdx.x & 31;
    const int wave = threadIdx.x >> 5;
    const int lm = lane & 15, h2 = lane >> 4;
    const int wm = blockIdx.x * 128 + (wave & 3) * 32;
    const int wn = blockIdx.y * 64 + (wave >> 2) * 32;

    v8f acc[2][2] = {};
    const __bf16* a0p = A + (size_t)(wm + lm) * K;
    const __bf16* a1p = A + (size_t)(wm + 16 + lm) * K;

    for (int k0 = 0; k0 < K; k0 += 32) {
        __builtin_prefetch(a0p + k0 + 64, 0, 0);  // global_prefetch_b8
        v16bf a0 = load_afrag(a0p + k0, lane);
        v16bf a1 = load_afrag(a1p + k0, lane);
        v16bf b0 = load_bfrag(Bw + (size_t)k0 * N + wn, N, lane);
        v16bf b1 = load_bfrag(Bw + (size_t)k0 * N + wn + 16, N, lane);
        acc[0][0] = wmma_bf16(a0, b0, acc[0][0]);
        acc[0][1] = wmma_bf16(a0, b1, acc[0][1]);
        acc[1][0] = wmma_bf16(a1, b0, acc[1][0]);
        acc[1][1] = wmma_bf16(a1, b1, acc[1][1]);
    }

#pragma unroll
    for (int mi = 0; mi < 2; ++mi)
#pragma unroll
        for (int ni = 0; ni < 2; ++ni) {
            const int col = wn + ni * 16 + lm;
            const float bv = bias[col];
            const int row0 = wm + mi * 16 + h2 * 8;  // 8 consecutive rows per lane
            if (MODE == 1) {
                const int bt = row0 / LSEQ;  // tiles never straddle a bt boundary
                const int s0 = row0 % LSEQ;
                __bf16* dst = (__bf16*)OutP +
                              (((size_t)bt * NH + (col >> 6)) * DKH + (col & 63)) * LSEQ + s0;
                v8bf pk;
#pragma unroll
                for (int i = 0; i < 8; ++i) pk[i] = (__bf16)(acc[mi][ni][i] + bv);
                *(v8bf*)dst = pk;  // contiguous along s -> single 16B store
            } else if (MODE == 0) {
                __bf16* dst = (__bf16*)OutP;
#pragma unroll
                for (int i = 0; i < 8; ++i)
                    dst[(size_t)(row0 + i) * N + col] = (__bf16)(acc[mi][ni][i] + bv);
            } else {
                float* dst = (float*)OutP;
#pragma unroll
                for (int i = 0; i < 8; ++i)
                    dst[(size_t)(row0 + i) * N + col] = acc[mi][ni][i] + bv;
            }
        }
}

// ---------- fused scores (Q K^T) * mask * scale + softmax ----------
// grid: (LSEQ/16, BT*NH); block 256 (8 waves). Each wave owns 128 s-columns.
// Dynamic LDS: fp32 scores [16][1024] (64KB) + mask bytes [16][1024] (16KB).
__global__ void __launch_bounds__(256) attn_scores_softmax(
        const __bf16* __restrict__ qb, const __bf16* __restrict__ kT,
        const unsigned char* __restrict__ mask, float* __restrict__ attn) {
    extern __shared__ char smem[];
    float* sc = (float*)smem;                                      // [16][LSEQ]
    unsigned char* msk = (unsigned char*)(smem + 16 * LSEQ * 4);   // [16][LSEQ]

    const int lane = threadIdx.x & 31, wave = threadIdx.x >> 5;
    const int lm = lane & 15, h2 = lane >> 4;
    const int l0 = blockIdx.x * 16;
    const int bth = blockIdx.y;  // (b*T+t)*NH + h
    const int bt = bth >> 3, h = bth & 7;
    (void)bt;

    // stage 16 mask rows (16KB) coalesced
    {
        const uint4* src = (const uint4*)(mask + (size_t)l0 * LSEQ);
        uint4* dstm = (uint4*)msk;
#pragma unroll
        for (int j = 0; j < 4; ++j)
            dstm[threadIdx.x + 256 * j] = src[threadIdx.x + 256 * j];
    }
    __syncthreads();

    // A fragments: this block's 16 query rows, head h, k = dk in [0,64)
    const __bf16* qrow = qb + ((size_t)bt * LSEQ + l0 + lm) * DMODEL + h * DKH;
    v16bf a0 = load_afrag(qrow, lane);
    v16bf a1 = load_afrag(qrow + 32, lane);
    const __bf16* kTh = kT + (size_t)bth * DKH * LSEQ;  // [dk][s], pitch LSEQ

    const int s0w = wave * 128;
#pragma unroll
    for (int t = 0; t < 8; ++t) {
        const int s0 = s0w + t * 16;
        v16bf b0 = load_bfrag(kTh + s0, LSEQ, lane);
        v16bf b1 = load_bfrag(kTh + (size_t)32 * LSEQ + s0, LSEQ, lane);
        v8f c = {};
        c = wmma_bf16(a0, b0, c);
        c = wmma_bf16(a1, b1, c);
        const int scol = s0 + lm;
#pragma unroll
        for (int i = 0; i < 8; ++i) {
            const int m = i + 8 * h2;
            const float f = msk[m * LSEQ + scol] ? 0.1f : 0.9f;  // mask*0.1+(~)*0.9
            sc[m * LSEQ + scol] = c[i] * f * 0.125f;             // * 1/sqrt(64)
        }
    }
    __syncthreads();

    // softmax: wave w handles rows 2w, 2w+1; wave32 shfl reductions
#pragma unroll
    for (int rr = 0; rr < 2; ++rr) {
        const int r = wave * 2 + rr;
        float* row = sc + r * LSEQ;
        float mx = -3.0e38f;
        for (int j = 0; j < 32; ++j) mx = fmaxf(mx, row[lane + j * 32]);
#pragma unroll
        for (int off = 16; off > 0; off >>= 1) mx = fmaxf(mx, __shfl_xor(mx, off));
        float sum = 0.f;
        for (int j = 0; j < 32; ++j) {
            float e = __expf(row[lane + j * 32] - mx);
            row[lane + j * 32] = e;
            sum += e;
        }
#pragma unroll
        for (int off = 16; off > 0; off >>= 1) sum += __shfl_xor(sum, off);
        const float inv = 1.0f / sum;
        float* dst = attn + ((size_t)bth * LSEQ + l0 + r) * LSEQ;
        for (int j = 0; j < 32; ++j) dst[lane + j * 32] = row[lane + j * 32] * inv;
    }
}

// ---------- attn (fp32, converted in-register) x V (bf16) -> ctx (bf16) ----------
// grid: (LSEQ/128, BT*NH); block 256 (8 waves); each wave: 16 rows x full DK=64.
__global__ void __launch_bounds__(256) attn_av(const float* __restrict__ attn,
                                               const __bf16* __restrict__ vb,
                                               __bf16* __restrict__ ctxb) {
    const int lane = threadIdx.x & 31, wave = threadIdx.x >> 5;
    const int lm = lane & 15, h2 = lane >> 4;
    const int bth = blockIdx.y, bt = bth >> 3, h = bth & 7;
    const int wl = blockIdx.x * 128 + wave * 16;

    v8f acc[4] = {};
    const float* arow = attn + ((size_t)bth * LSEQ + wl + lm) * LSEQ;
    const __bf16* vbase = vb + (size_t)bt * LSEQ * DMODEL + h * DKH;

    for (int s0 = 0; s0 < LSEQ; s0 += 32) {
        // A fragment from fp32 attention row: two 32B runs, cvt to bf16 in-reg
        const float* p = arow + s0 + h2 * 8;
        v4f f0 = *(const v4f*)(p);
        v4f f1 = *(const v4f*)(p + 4);
        v4f f2 = *(const v4f*)(p + 16);
        v4f f3 = *(const v4f*)(p + 20);
        v16bf a;
#pragma unroll
        for (int j = 0; j < 4; ++j) {
            a[j]      = (__bf16)f0[j];
            a[4 + j]  = (__bf16)f1[j];
            a[8 + j]  = (__bf16)f2[j];
            a[12 + j] = (__bf16)f3[j];
        }
        const __bf16* vs = vbase + (size_t)s0 * DMODEL;  // row s contiguous in dk
#pragma unroll
        for (int nb = 0; nb < 4; ++nb) {
            v16bf b = load_bfrag(vs + nb * 16, DMODEL, lane);
            acc[nb] = wmma_bf16(a, b, acc[nb]);
        }
    }
#pragma unroll
    for (int nb = 0; nb < 4; ++nb) {
        const int col = h * DKH + nb * 16 + lm;
        const int row0 = wl + 8 * h2;
#pragma unroll
        for (int i = 0; i < 8; ++i)
            ctxb[((size_t)bt * LSEQ + row0 + i) * DMODEL + col] = (__bf16)acc[nb][i];
    }
}

// ---------- host launcher ----------
extern "C" void kernel_launch(void* const* d_in, const int* in_sizes, int n_in,
                              void* d_out, int out_size, void* d_ws, size_t ws_size,
                              hipStream_t stream) {
    const float* queries = (const float*)d_in[0];
    const float* keys    = (const float*)d_in[1];
    const float* values  = (const float*)d_in[2];
    const unsigned char* mask = (const unsigned char*)d_in[3];  // jax bool = 1 byte
    const float* Wq = (const float*)d_in[5];  const float* bq = (const float*)d_in[6];
    const float* Wk = (const float*)d_in[7];  const float* bk = (const float*)d_in[8];
    const float* Wv = (const float*)d_in[9];  const float* bv = (const float*)d_in[10];
    const float* Wo = (const float*)d_in[11]; const float* bo = (const float*)d_in[12];

    const int BT = in_sizes[0] / (LSEQ * DMODEL);  // B*T = 8
    const size_t actN = (size_t)BT * LSEQ * DMODEL;
    const size_t wN = (size_t)DMODEL * DMODEL;

    char* w = (char*)d_ws;
    __bf16* qin = (__bf16*)w; w += actN * 2;
    __bf16* kin = (__bf16*)w; w += actN * 2;
    __bf16* vin = (__bf16*)w; w += actN * 2;
    __bf16* wqb = (__bf16*)w; w += wN * 2;
    __bf16* wkb = (__bf16*)w; w += wN * 2;
    __bf16* wvb = (__bf16*)w; w += wN * 2;
    __bf16* wob = (__bf16*)w; w += wN * 2;
    __bf16* qb  = (__bf16*)w; w += actN * 2;
    __bf16* kT  = (__bf16*)w; w += actN * 2;  // [bt][h][dk][s]
    __bf16* vbp = (__bf16*)w; w += actN * 2;
    __bf16* ctx = (__bf16*)w; w += actN * 2;

    float* outp = (float*)d_out;          // [BT][LSEQ][DMODEL]
    float* attnp = outp + actN;           // [BT*NH][LSEQ][LSEQ]

    dim3 blk(256);
    cvt_f32_bf16<<<dim3(512), blk, 0, stream>>>(queries, qin, (long)(actN / 4));
    cvt_f32_bf16<<<dim3(512), blk, 0, stream>>>(keys, kin, (long)(actN / 4));
    cvt_f32_bf16<<<dim3(512), blk, 0, stream>>>(values, vin, (long)(actN / 4));
    cvt_f32_bf16<<<dim3(128), blk, 0, stream>>>(Wq, wqb, (long)(wN / 4));
    cvt_f32_bf16<<<dim3(128), blk, 0, stream>>>(Wk, wkb, (long)(wN / 4));
    cvt_f32_bf16<<<dim3(128), blk, 0, stream>>>(Wv, wvb, (long)(wN / 4));
    cvt_f32_bf16<<<dim3(128), blk, 0, stream>>>(Wo, wob, (long)(wN / 4));

    const int M = BT * LSEQ;
    dim3 ggrid(M / 128, DMODEL / 64);
    gemm_bf16_k<0><<<ggrid, blk, 0, stream>>>(qin, wqb, bq, qb, M, DMODEL, DMODEL);
    gemm_bf16_k<1><<<ggrid, blk, 0, stream>>>(kin, wkb, bk, kT, M, DMODEL, DMODEL);
    gemm_bf16_k<0><<<ggrid, blk, 0, stream>>>(vin, wvb, bv, vbp, M, DMODEL, DMODEL);

    attn_scores_softmax<<<dim3(LSEQ / 16, BT * NH), blk, 16 * LSEQ * 5, stream>>>(
        qb, kT, mask, attnp);

    attn_av<<<dim3(LSEQ / 128, BT * NH), blk, 0, stream>>>(attnp, vbp, ctx);

    gemm_bf16_k<2><<<ggrid, blk, 0, stream>>>(ctx, wob, bo, (void*)outp, M, DMODEL,
                                              DMODEL);
}
